// CSSAM_27745488733025
// MI455X (gfx1250) — matrix operand order repaired and checked by reading, not compiled, not run on hardware
//
#include <hip/hip_runtime.h>

// ---------------------------------------------------------------------------
// CDNA5 (gfx1250) implementation of CSSAM attention block.
// wave32, WMMA f32_16x16x32_bf16 everywhere, flash-attention streaming softmax.
// Round 2: V stored transposed (contiguous B-fragments), 2 query tiles / wave.
// ---------------------------------------------------------------------------

typedef __attribute__((ext_vector_type(16))) __bf16 v16bf;
typedef __attribute__((ext_vector_type(8)))  float  v8f;

#define BATCH  2
#define CDIM   256
#define L_Q    4096      // 64*64
#define L_KV   4096      // 64*64 output of stride-2 unfold of 128x128
#define NHEAD  8
#define HDIM   32
#define MROWS  (BATCH * L_Q)   // 8192 rows for all projections

__device__ __forceinline__ __bf16 f2bf(float f) {
  union { float f; unsigned u; } x; x.f = f;
  unsigned r = x.u + 0x7FFFu + ((x.u >> 16) & 1u);   // round-to-nearest-even
  unsigned short h = (unsigned short)(r >> 16);
  return __builtin_bit_cast(__bf16, h);
}

// ---------------------------------------------------------------------------
// f32 -> bf16 convert (weights)
// ---------------------------------------------------------------------------
__global__ void cvt_f32_bf16(const float* __restrict__ in, __bf16* __restrict__ out, int n) {
  int i = blockIdx.x * blockDim.x + threadIdx.x;
  if (i < n) out[i] = f2bf(in[i]);
}

// ---------------------------------------------------------------------------
// query[b, l, c] = src[b, c, l]  (1x1 unfold == spatial flatten + transpose)
// ---------------------------------------------------------------------------
__global__ void build_query_bf16(const float* __restrict__ src, __bf16* __restrict__ q) {
  int i = blockIdx.x * blockDim.x + threadIdx.x;
  if (i >= BATCH * L_Q * CDIM) return;
  int c = i % CDIM;
  int l = (i / CDIM) % L_Q;
  int b = i / (CDIM * L_Q);
  q[i] = f2bf(src[((size_t)b * CDIM + c) * L_Q + l]);
}

// ---------------------------------------------------------------------------
// kv[b, l, j]: 3x3/s2/p1 im2col of feat keeping first 256 of C*9 channels.
// j -> channel c=j/9, kernel offset kh=(j%9)/3, kw=j%3.
// ---------------------------------------------------------------------------
__global__ void build_kv_bf16(const float* __restrict__ feat, __bf16* __restrict__ kv) {
  int i = blockIdx.x * blockDim.x + threadIdx.x;
  if (i >= BATCH * L_KV * CDIM) return;
  int j = i % CDIM;
  int l = (i / CDIM) % L_KV;
  int b = i / (CDIM * L_KV);
  int c  = j / 9;
  int k9 = j % 9;
  int kh = k9 / 3, kw = k9 % 3;
  int oh = l / 64, ow = l % 64;
  int ih = oh * 2 + kh - 1;
  int iw = ow * 2 + kw - 1;
  float v = 0.0f;
  if (ih >= 0 && ih < 128 && iw >= 0 && iw < 128)
    v = feat[(((size_t)b * CDIM + c) * 128 + ih) * 128 + iw];
  kv[i] = f2bf(v);
}

// ---------------------------------------------------------------------------
// WMMA GEMM: out[m,n] = (sum_k A[m,k] * W[n,k] + bias[n]) * alpha
// A: [MROWS, 256] bf16 row-major; W: [256, 256] bf16 (W[n,k] row-major).
// mode 0: store bf16 row-major [M, C] to outb.
// mode 1: out-projection epilogue — fold to [B,C,64,64], multiply by src, f32.
// mode 2: store bf16 TRANSPOSED [B, C, L] to outb (for V: key-contiguous).
// One 16x16 output tile per wave; 8 waves / block.
// ---------------------------------------------------------------------------
__global__ void gemm_proj_wmma(const __bf16* __restrict__ A, const __bf16* __restrict__ W,
                               const float* __restrict__ bias, float alpha,
                               __bf16* __restrict__ outb,
                               const float* __restrict__ srcf, float* __restrict__ outf,
                               int mode) {
  int wave = threadIdx.x >> 5;
  int lane = threadIdx.x & 31;
  int tile = blockIdx.x * 8 + wave;          // (MROWS/16) * (CDIM/16) tiles
  int mt = tile >> 4;                        // CDIM/16 == 16 n-tiles
  int nt = tile & 15;
  int m0 = mt * 16, n0 = nt * 16;
  int row  = lane & 15;                      // A row / B column within tile
  int half = lane >> 4;

  const __bf16* Arow = A + (size_t)(m0 + row) * CDIM;
  const __bf16* Wn   = W + (size_t)(n0 + row) * CDIM;

  v8f acc = {};
#pragma unroll
  for (int kk = 0; kk < CDIM; kk += 32) {
    v16bf a, bfr;
    const __bf16* pa0 = Arow + kk + half * 16;       // K = kk + half*16 + e
    const __bf16* pa1 = Arow + kk + half * 16 + 8;   // K = kk + half*16 + 8 + e
#pragma unroll
    for (int e = 0; e < 8; e++) { a[e] = pa0[e]; a[e + 8] = pa1[e]; }
    const __bf16* pb = Wn + kk + half * 16;          // K = kk + half*16 + e
#pragma unroll
    for (int e = 0; e < 16; e++) bfr[e] = pb[e];
    acc = __builtin_amdgcn_wmma_f32_16x16x32_bf16(false, a, false, bfr,
                                                  (short)0, acc, false, false);
  }

  int n = n0 + row;
  float bn = bias[n];
#pragma unroll
  for (int r = 0; r < 8; r++) {
    int m = m0 + r + 8 * half;
    float val = (acc[r] + bn) * alpha;
    int b = m >> 12;                 // m / 4096
    int l = m & 4095;
    if (mode == 0) {
      outb[(size_t)m * CDIM + n] = f2bf(val);
    } else if (mode == 2) {
      // transposed: [B, C, L] -> per-(head,dim) rows contiguous over keys
      outb[((size_t)b * CDIM + n) * L_KV + l] = f2bf(val);
    } else {
      size_t idx = ((size_t)b * CDIM + n) * L_Q + l;   // [B, C, 64, 64] layout
      outf[idx] = val * srcf[idx];
    }
  }
}

// ---------------------------------------------------------------------------
// Flash attention: one wave handles one (b, h, 32-query block) == 2 Q-tiles,
// sharing K/V fragments between the two tiles (8 WMMAs per 32-key chunk).
// K is [B, L, C] (dim-contiguous B-frags); V is transposed [B, C, L]
// (key-contiguous B-frags). Online softmax in f32; P relayout via LDS.
// ---------------------------------------------------------------------------
__global__ void flash_attn_wmma(const __bf16* __restrict__ Q, const __bf16* __restrict__ K,
                                const __bf16* __restrict__ VT, __bf16* __restrict__ O) {
  __shared__ float ldsP[8 * 16 * 32];                 // 16 KB: one 16x32 tile / wave
  int wave = threadIdx.x >> 5;
  int lane = threadIdx.x & 31;
  int task = blockIdx.x * 8 + wave;                   // B*NHEAD*(L_Q/32) tasks
  int b   = task / (NHEAD * (L_Q / 32));
  int rem = task % (NHEAD * (L_Q / 32));
  int h   = rem / (L_Q / 32);
  int q0  = (rem % (L_Q / 32)) * 32;
  int row  = lane & 15;
  int half = lane >> 4;
  float* myP = ldsP + wave * (16 * 32);

  // ---- Q fragments (A layout, 16 queries x 32 head-dims), 2 tiles ----
  v16bf qa[2];
#pragma unroll
  for (int t = 0; t < 2; t++) {
    const __bf16* qp  = Q + ((size_t)(b * L_Q + q0 + t * 16 + row)) * CDIM + h * HDIM;
    const __bf16* qp0 = qp + half * 16;
    const __bf16* qp1 = qp + half * 16 + 8;
#pragma unroll
    for (int e = 0; e < 8; e++) { qa[t][e] = qp0[e]; qa[t][e + 8] = qp1[e]; }
  }

  float mrun[2][8], lrun[2][8];
  v8f oacc[2][2];
#pragma unroll
  for (int t = 0; t < 2; t++) {
    oacc[t][0] = (v8f){};
    oacc[t][1] = (v8f){};
#pragma unroll
    for (int r = 0; r < 8; r++) { mrun[t][r] = -3.0e38f; lrun[t][r] = 0.0f; }
  }

  const __bf16* Kb  = K  + (size_t)b * L_KV * CDIM + h * HDIM;
  const __bf16* VTb = VT + ((size_t)b * CDIM + h * HDIM) * L_KV;

  for (int kc = 0; kc < L_KV; kc += 32) {
    // ---- K^T fragments: B layout, n = key (lane%16), K = dim (half*16 + e) ----
    v16bf kf0, kf1;
    const __bf16* kp0 = Kb + (size_t)(kc + row) * CDIM + half * 16;
    const __bf16* kp1 = kp0 + (size_t)16 * CDIM;
#pragma unroll
    for (int e = 0; e < 16; e++) { kf0[e] = kp0[e]; kf1[e] = kp1[e]; }

    // ---- V fragments: B layout, n = head-dim (lane%16), K = key (half*16+e),
    //      contiguous over keys thanks to transposed V storage ----
    v16bf vf0, vf1;
    const __bf16* vp0 = VTb + (size_t)row * L_KV + kc + half * 16;        // dims 0..15
    const __bf16* vp1 = vp0 + (size_t)16 * L_KV;                          // dims 16..31
#pragma unroll
    for (int e = 0; e < 16; e++) { vf0[e] = vp0[e]; vf1[e] = vp1[e]; }

    if (kc + 32 < L_KV) {
      __builtin_prefetch(kp0 + (size_t)32 * CDIM, 0, 1);   // global_prefetch_b8
      __builtin_prefetch(kp1 + (size_t)32 * CDIM, 0, 1);
      __builtin_prefetch(vp0 + 32, 0, 1);
      __builtin_prefetch(vp1 + 32, 0, 1);
    }

#pragma unroll
    for (int t = 0; t < 2; t++) {
      v8f z = {};
      v8f s0 = __builtin_amdgcn_wmma_f32_16x16x32_bf16(false, qa[t], false, kf0,
                                                       (short)0, z, false, false);
      v8f s1 = __builtin_amdgcn_wmma_f32_16x16x32_bf16(false, qa[t], false, kf1,
                                                       (short)0, z, false, false);

      // ---- online softmax (row = r + 8*half, replicated across 16 lanes) ----
      float p0[8], p1[8];
#pragma unroll
      for (int r = 0; r < 8; r++) {
        float cmax = fmaxf(s0[r], s1[r]);
#pragma unroll
        for (int msk = 1; msk < 16; msk <<= 1)
          cmax = fmaxf(cmax, __shfl_xor(cmax, msk, 32));
        float mnew = fmaxf(mrun[t][r], cmax);
        float cf   = __expf(mrun[t][r] - mnew);
        p0[r] = __expf(s0[r] - mnew);
        p1[r] = __expf(s1[r] - mnew);
        float rs = p0[r] + p1[r];
#pragma unroll
        for (int msk = 1; msk < 16; msk <<= 1)
          rs += __shfl_xor(rs, msk, 32);
        lrun[t][r] = lrun[t][r] * cf + rs;
        mrun[t][r] = mnew;
        oacc[t][0][r] *= cf;
        oacc[t][1][r] *= cf;
        int M = r + 8 * half;
        myP[M * 32 + row]      = p0[r];   // C-layout -> LDS row-major tile
        myP[M * 32 + 16 + row] = p1[r];
      }
      asm volatile("s_wait_dscnt 0" ::: "memory");   // same-wave LDS RAW fence

      // ---- P as A fragment (16 queries x 32 keys) read back from LDS ----
      v16bf pa;
      const float* pr = myP + row * 32;
#pragma unroll
      for (int e = 0; e < 8; e++) {
        pa[e]     = f2bf(pr[half * 16 + e]);
        pa[e + 8] = f2bf(pr[half * 16 + 8 + e]);
      }

      oacc[t][0] = __builtin_amdgcn_wmma_f32_16x16x32_bf16(false, pa, false, vf0,
                                                           (short)0, oacc[t][0], false, false);
      oacc[t][1] = __builtin_amdgcn_wmma_f32_16x16x32_bf16(false, pa, false, vf1,
                                                           (short)0, oacc[t][1], false, false);
    }
  }

  // ---- epilogue: normalize and store bf16 attention output [B, Lq, C] ----
#pragma unroll
  for (int t = 0; t < 2; t++) {
#pragma unroll
    for (int r = 0; r < 8; r++) {
      float inv = 1.0f / lrun[t][r];
      int qi = q0 + t * 16 + r + 8 * half;
      size_t obase = ((size_t)(b * L_Q + qi)) * CDIM + h * HDIM + row;
      O[obase]      = f2bf(oacc[t][0][r] * inv);
      O[obase + 16] = f2bf(oacc[t][1][r] * inv);
    }
  }
}

// ---------------------------------------------------------------------------
extern "C" void kernel_launch(void* const* d_in, const int* in_sizes, int n_in,
                              void* d_out, int out_size, void* d_ws, size_t ws_size,
                              hipStream_t stream) {
  (void)in_sizes; (void)n_in; (void)out_size; (void)ws_size;
  const float* feat = (const float*)d_in[0];
  const float* src  = (const float*)d_in[1];
  const float* Wq   = (const float*)d_in[2];
  const float* bq   = (const float*)d_in[3];
  const float* Wk   = (const float*)d_in[4];
  const float* bk   = (const float*)d_in[5];
  const float* Wv   = (const float*)d_in[6];
  const float* bv   = (const float*)d_in[7];
  const float* Wo   = (const float*)d_in[8];
  const float* bo   = (const float*)d_in[9];
  float* out = (float*)d_out;

  char* ws = (char*)d_ws;
  size_t off = 0;
  auto take = [&](size_t bytes) -> char* {
    char* p = ws + off;
    off += (bytes + 255) & ~(size_t)255;
    return p;
  };
  const size_t actB = (size_t)MROWS * CDIM * sizeof(__bf16);   // 4 MB each
  const size_t wB   = (size_t)CDIM * CDIM * sizeof(__bf16);    // 128 KB each
  __bf16* kvb    = (__bf16*)take(actB);
  __bf16* queryb = (__bf16*)take(actB);
  __bf16* Wqb    = (__bf16*)take(wB);
  __bf16* Wkb    = (__bf16*)take(wB);
  __bf16* Wvb    = (__bf16*)take(wB);
  __bf16* Wob    = (__bf16*)take(wB);
  __bf16* qb     = (__bf16*)take(actB);
  __bf16* kb     = (__bf16*)take(actB);
  __bf16* vtb    = (__bf16*)take(actB);   // V stored transposed [B, C, L]
  __bf16* ob     = (__bf16*)take(actB);

  const int nW = CDIM * CDIM;
  cvt_f32_bf16<<<(nW + 255) / 256, 256, 0, stream>>>(Wq, Wqb, nW);
  cvt_f32_bf16<<<(nW + 255) / 256, 256, 0, stream>>>(Wk, Wkb, nW);
  cvt_f32_bf16<<<(nW + 255) / 256, 256, 0, stream>>>(Wv, Wvb, nW);
  cvt_f32_bf16<<<(nW + 255) / 256, 256, 0, stream>>>(Wo, Wob, nW);

  const int nAct = MROWS * CDIM;
  build_query_bf16<<<(nAct + 255) / 256, 256, 0, stream>>>(src, queryb);
  build_kv_bf16<<<(nAct + 255) / 256, 256, 0, stream>>>(feat, kvb);

  // q carries the 1/sqrt(HD) attention scale (applied after bias, as in torch)
  const float qscale = 0.17677669529663687f;   // 1/sqrt(32)
  gemm_proj_wmma<<<1024, 256, 0, stream>>>(queryb, Wqb, bq, qscale, qb,  nullptr, nullptr, 0);
  gemm_proj_wmma<<<1024, 256, 0, stream>>>(kvb,    Wkb, bk, 1.0f,   kb,  nullptr, nullptr, 0);
  gemm_proj_wmma<<<1024, 256, 0, stream>>>(kvb,    Wvb, bv, 1.0f,   vtb, nullptr, nullptr, 2);

  // 2 Q-tiles per wave: B*NHEAD*(L_Q/32) = 2048 waves -> 256 blocks of 8 waves
  flash_attn_wmma<<<256, 256, 0, stream>>>(qb, kb, vtb, ob);

  // out projection fused with fold + elementwise multiply by src -> f32 d_out
  gemm_proj_wmma<<<1024, 256, 0, stream>>>(ob, Wob, bo, 1.0f, nullptr, src, out, 1);
}
